// SingleLayerElmanRNN_28003186770565
// MI455X (gfx1250) — matrix operand order repaired and checked by reading, not compiled, not run on hardware
//
#include <hip/hip_runtime.h>
#include <hip/hip_bf16.h>
#include <cstdint>

// Problem constants (from reference): B=64, T=4096, I=128, H=128
#define B_  64
#define T_  4096
#define I_  128
#define H_  128
// Padded LDS row stride (in bf16 elements): 136*2B = 272B = 68 DWORDs.
// 68 mod 64 == 4 -> the 16 per-lane ds_load_b128 A-fragment fetches hit
// disjoint 4-bank groups (conflict-free) instead of a 16-way conflict.
#define STRIDE_ 136

typedef __attribute__((ext_vector_type(16))) __bf16 v16bf;
typedef __attribute__((ext_vector_type(8)))  __bf16 v8bf;
typedef __attribute__((ext_vector_type(8)))  float  v8f;

// tanh via the hardware transcendentals (v_exp_f32 + v_rcp_f32):
// tanh(x) = sign(x) * (1 - 2/(exp(2|x|)+1));  exp(2a) = exp2(a * 2*log2(e))
__device__ __forceinline__ float fast_tanh(float x) {
  float ax = __builtin_fabsf(x);
  float e  = __builtin_amdgcn_exp2f(ax * 2.88539008177792681472f); // 2/ln(2)
  float r  = 1.0f - 2.0f * __builtin_amdgcn_rcpf(e + 1.0f);        // inf-safe: ->1
  return __builtin_copysignf(r, x);
}

// One workgroup per 16-row batch tile (4 blocks — the scan dependency caps
// usable parallelism). 8 waves; wave w owns output columns [16w, 16w+16).
// min-blocks=1: don't throttle registers for occupancy we can't use anyway.
__global__ __launch_bounds__(256, 1) void elman_rnn_scan(
    const float* __restrict__ X,     // (B, T, I)
    const float* __restrict__ W,     // (I+H, H) row-major; rows 0..I-1 = W_x
    const float* __restrict__ bias,  // (H)
    float* __restrict__ out)         // (B, T, H)
{
  __shared__ __bf16 xbuf[2][16 * STRIDE_];  // staged x_t tile (double buffer)
  __shared__ __bf16 hbuf[2][16 * STRIDE_];  // h_{t-1} tile   (double buffer)

  const int tid  = threadIdx.x;
  const int lane = tid & 31;
  const int wave = tid >> 5;               // 0..7 -> N tile
  const int n0   = wave * 16;
  const int nl   = lane & 15;
  const int kbB  = (lane < 16) ? 0 : 16;   // B-fragment K offset (half-wave)
  const int kbA  = (lane < 16) ? 0 : 16;   // A-fragment K offset (half-wave)
  const int rb   = blockIdx.x * 16;        // batch row base
  const int m0   = (lane < 16) ? 0 : 8;    // C-fragment row base

  // ---- Pack W_x / W_h into resident WMMA B-fragments (64 VGPRs total).
  // B layout (32x16 bf16): lane -> N = lane&15; VGPR j holds K pair
  // {kbB+2j, kbB+2j+1} of this fragment's 32-wide K slice.
  v16bf Bx[4], Bh[4];
  #pragma unroll
  for (int f = 0; f < 4; ++f) {
    #pragma unroll
    for (int j = 0; j < 8; ++j) {
      int k = 32 * f + kbB + 2 * j;
      Bx[f][2*j]   = (__bf16)W[(size_t)(k)          * H_ + n0 + nl];
      Bx[f][2*j+1] = (__bf16)W[(size_t)(k + 1)      * H_ + n0 + nl];
      Bh[f][2*j]   = (__bf16)W[(size_t)(I_ + k)     * H_ + n0 + nl];
      Bh[f][2*j+1] = (__bf16)W[(size_t)(I_ + k + 1) * H_ + n0 + nl];
    }
  }
  const float bias_n = bias[n0 + nl];

  // Integer element offsets (keep all LDS accesses as direct __shared__
  // indexing so they lower to ds_load/ds_store, never flat_*).
  const int afrag = nl * STRIDE_ + kbA;      // A-fragment base (load side)
  const int hsto  = m0 * STRIDE_ + n0 + nl;  // h store base (C layout side)

  // out base for this lane's C column: rows (rb+m0+r), col n0+nl.
  float* outp = out + ((size_t)(rb + m0) * T_) * H_ + n0 + nl;

  // ---- Prologue: h_0 = 0; stage x_0 into LDS.
  const int xr_row  = tid >> 4;   // 0..15: which batch row this thread stages
  const int xr_part = tid & 15;   // 8 consecutive floats per thread
  const int xsto    = xr_row * STRIDE_ + xr_part * 8;
  const float* xsrc = X + ((size_t)(rb + xr_row) * T_) * I_ + xr_part * 8;
  for (int i = tid; i < 16 * STRIDE_; i += 256)
    hbuf[0][i] = (__bf16)0.0f;
  {
    v8bf v;
    #pragma unroll
    for (int j = 0; j < 8; ++j) v[j] = (__bf16)xsrc[j];
    *(v8bf*)&xbuf[0][xsto] = v;
  }
  __syncthreads();

  // ---- Sequential scan: one barrier per step (double-buffered LDS).
  #pragma unroll 1
  for (int t = 0; t < T_; ++t) {
    const int cur = t & 1;
    const int nxt = cur ^ 1;
    // Clamped prefetch index: no per-iteration branch / exec-mask traffic;
    // the duplicated final load+commit is harmless.
    const int tn = (t + 1 < T_) ? (t + 1) : (T_ - 1);

    // Prefetch x_{t+1} (independent of the recurrence -> hides HBM latency).
    float xr[8];
    {
      const float* src = xsrc + (size_t)tn * I_;
      #pragma unroll
      for (int j = 0; j < 8; ++j) xr[j] = src[j];
    }

    // Issue ALL 16 ds_load_b128 before any WMMA. A-fragment layout
    // (16x32 bf16): lane -> M = lane&15; VGPRs 0-3 = K kbA..kbA+7,
    // VGPRs 4-7 = K kbA+8..kbA+15 -> two 16B loads per fragment.
    v16bf ax[4], ah[4];
    #pragma unroll
    for (int f = 0; f < 4; ++f) {
      ((v8bf*)&ax[f])[0] = *(const v8bf*)&xbuf[cur][afrag + 32 * f];
      ((v8bf*)&ax[f])[1] = *(const v8bf*)&xbuf[cur][afrag + 32 * f + 8];
    }
    #pragma unroll
    for (int f = 0; f < 4; ++f) {
      ((v8bf*)&ah[f])[0] = *(const v8bf*)&hbuf[cur][afrag + 32 * f];
      ((v8bf*)&ah[f])[1] = *(const v8bf*)&hbuf[cur][afrag + 32 * f + 8];
    }
    // Hard scheduling fence: nothing crosses. Forces all 8 fragments to be
    // live at once -> RA cannot coalesce them into one octet -> the LDS
    // batch pipelines with partial dscnt waits instead of 8 full stalls.
    __builtin_amdgcn_sched_barrier(0);

    // C = bias, then 8 chained WMMAs. x-part first (its data was staged a
    // full step ago); h-part last so its loads hide behind the x WMMAs.
    v8f c;
    #pragma unroll
    for (int r = 0; r < 8; ++r) c[r] = bias_n;
    #pragma unroll
    for (int f = 0; f < 4; ++f)
      c = __builtin_amdgcn_wmma_f32_16x16x32_bf16(
              false, ax[f], false, Bx[f], (short)0, c, false, false);
    #pragma unroll
    for (int f = 0; f < 4; ++f)
      c = __builtin_amdgcn_wmma_f32_16x16x32_bf16(
              false, ah[f], false, Bh[f], (short)0, c, false, false);

    // tanh (v_exp_f32/v_rcp_f32), then stage h_t first (the next step's
    // barrier waits on these LDS stores) before the fire-and-forget
    // global stores of the output.
    float h[8];
    #pragma unroll
    for (int r = 0; r < 8; ++r) h[r] = fast_tanh(c[r]);
    #pragma unroll
    for (int r = 0; r < 8; ++r)
      hbuf[nxt][hsto + r * STRIDE_] = (__bf16)h[r];
    #pragma unroll
    for (int r = 0; r < 8; ++r)
      outp[(size_t)r * (T_ * H_) + (size_t)t * H_] = h[r];

    // Commit prefetched x_{t+1} (converted to bf16) into the next buffer.
    {
      v8bf v;
      #pragma unroll
      for (int j = 0; j < 8; ++j) v[j] = (__bf16)xr[j];
      *(v8bf*)&xbuf[nxt][xsto] = v;
    }
    __syncthreads();
  }
}

extern "C" void kernel_launch(void* const* d_in, const int* in_sizes, int n_in,
                              void* d_out, int out_size, void* d_ws, size_t ws_size,
                              hipStream_t stream) {
  (void)in_sizes; (void)n_in; (void)d_ws; (void)ws_size; (void)out_size;
  const float* X    = (const float*)d_in[0];   // (B, T, I) fp32
  const float* W    = (const float*)d_in[1];   // (I+H, H) fp32
  const float* bias = (const float*)d_in[2];   // (H) fp32
  float* out        = (float*)d_out;           // (B,T,H) then (1,T,H)

  elman_rnn_scan<<<dim3(4), dim3(256), 0, stream>>>(X, W, bias, out);

  // Second output = out[-1:, :, :] = contiguous batch-row 63 of `out`.
  hipMemcpyAsync(out + (size_t)B_ * T_ * H_,
                 out + (size_t)(B_ - 1) * T_ * H_,
                 (size_t)T_ * H_ * sizeof(float),
                 hipMemcpyDeviceToDevice, stream);
}
